// RNNModel_37555194036798
// MI455X (gfx1250) — matrix operand order repaired and checked by reading, not compiled, not run on hardware
//
#include <hip/hip_runtime.h>

#define B_ 512
#define T_ 1024
#define H_ 128
#define G_ 384   // 3H
#define O_ 4

#define BT 16          // batch tile per workgroup
#define NT_PER_WAVE 3  // r/z/n blocks: wave w owns n-tiles {w, w+8, w+16}
#define KSTEPS 4       // 128 / 32
#define HP 136         // padded pitch (halves) for h16: stride 68 dwords -> full bank spread

typedef _Float16 v16h __attribute__((ext_vector_type(16)));
typedef _Float16 v8h  __attribute__((ext_vector_type(8)));
typedef float    v8f  __attribute__((ext_vector_type(8)));
typedef float    v4f  __attribute__((ext_vector_type(4)));

// Fast sigmoid/tanh from hardware v_exp_f32 + v_rcp_f32 (no IEEE div expansion).
__device__ __forceinline__ float sigmoidf_(float x) {
    return __builtin_amdgcn_rcpf(1.0f + __expf(-x));
}
// tanh(x) = 1 - 2/(1 + exp(2x)); overflow-safe: exp->inf => +1, exp->0 => -1.
__device__ __forceinline__ float tanhf_(float x) {
    return 1.0f - 2.0f * __builtin_amdgcn_rcpf(1.0f + __expf(2.0f * x));
}

__global__ __launch_bounds__(256, 1)
void gru_persistent_wmma(const float* __restrict__ x,
                         const float* __restrict__ w_ih,
                         const float* __restrict__ w_hh,
                         const float* __restrict__ b_ih,
                         const float* __restrict__ b_hh,
                         const float* __restrict__ fc_w,
                         const float* __restrict__ fc_b,
                         float* __restrict__ out)
{
    __shared__ _Float16 s_h16[BT * HP] __attribute__((aligned(16))); // f16 A-operand copy of h
    __shared__ float    s_x[BT]        __attribute__((aligned(16))); // x[:,t] for this batch tile
    __shared__ float    s_h[BT * H_];                                // final h (for FC stage only)

    const int tid  = threadIdx.x;
    const int lane = tid & 31;
    const int wave = tid >> 5;
    const int b0   = blockIdx.x * BT;

    // ---- zero the f16 hidden-state copy ----
    for (int i = tid; i < BT * HP; i += 256) s_h16[i] = (_Float16)0.0f;

    // Wave w owns gate columns j = 16w + (lane&15) across all three gate blocks:
    //   acc[p] covers g = 16w + 128p + (lane&15)  ->  p=0:r, p=1:z, p=2:n
    const int jl = wave * 16 + (lane & 15);

    // ---- per-lane loop-invariant gate parameters (registers) ----
    const float wr  = w_ih[jl],          wz  = w_ih[H_ + jl],      wn  = w_ih[2 * H_ + jl];
    const float br  = b_ih[jl],          bz  = b_ih[H_ + jl],      bn  = b_ih[2 * H_ + jl];
    const float bhr = b_hh[jl],          bhz = b_hh[H_ + jl],      bhn = b_hh[2 * H_ + jl];

    // ---- preload loop-invariant B fragments (w_hh, f16) into registers ----
    // B 32x16 layout: lane&15 = column N (gate g); lanes 0-15 hold K 0..15,
    // lanes 16-31 hold K 16..31 as 16 consecutive halves.
    v16h bfrag[NT_PER_WAVE][KSTEPS];
    {
        const int n  = lane & 15;
        const int kb = (lane >> 4) * 16;
        for (int p = 0; p < NT_PER_WAVE; ++p) {
            const int g = (wave + 8 * p) * 16 + n;   // = 16*wave + 128*p + n
            for (int ks = 0; ks < KSTEPS; ++ks) {
                const float* wp = w_hh + g * H_ + ks * 32 + kb;
                v16h bf;
                #pragma unroll
                for (int i = 0; i < 16; ++i) bf[i] = (_Float16)wp[i];
                bfrag[p][ks] = bf;
            }
        }
    }

    // A 16x32 layout: lane&15 = row M; ko = 16*(lane>=16);
    // halves 0..15 = K ko..ko+15 (contiguous, matching B).
    const int m  = lane & 15;
    const int ko = (lane >> 4) * 16;
    const int mo = (lane >> 4) * 8;      // C/D-tile row offset for this lane

    // register-resident hidden state: hreg[i] = h[batch mo+i][jl]
    float hreg[8];
    #pragma unroll
    for (int i = 0; i < 8; ++i) hreg[i] = 0.0f;

    __syncthreads();

    for (int t = 0; t < T_; ++t) {
        if (tid < BT) s_x[tid] = x[(b0 + tid) * T_ + t];

        // ---- load A fragments (h_f16) for all 4 K-steps ----
        v16h afrag[KSTEPS];
        #pragma unroll
        for (int ks = 0; ks < KSTEPS; ++ks) {
            const _Float16* ap = &s_h16[m * HP + ks * 32 + ko];
            v8h lo = *(const v8h*)ap;
            v8h hi = *(const v8h*)(ap + 8);
            v16h a;
            #pragma unroll
            for (int i = 0; i < 8; ++i) { a[i] = lo[i]; a[8 + i] = hi[i]; }
            afrag[ks] = a;
        }
        __syncthreads();   // all waves have read h16 (and s_x is visible); safe to overwrite below

        // ---- gh = h @ w_hh^T via WMMA (12 wmma per wave per step) ----
        v8f acc[NT_PER_WAVE];
        #pragma unroll
        for (int p = 0; p < NT_PER_WAVE; ++p)
            #pragma unroll
            for (int i = 0; i < 8; ++i) acc[p][i] = 0.0f;

        #pragma unroll
        for (int ks = 0; ks < KSTEPS; ++ks)
            #pragma unroll
            for (int p = 0; p < NT_PER_WAVE; ++p)
                acc[p] = __builtin_amdgcn_wmma_f32_16x16x32_f16(
                    false, afrag[ks], false, bfrag[p][ks], (short)0, acc[p],
                    false, false);

        // ---- broadcast x values for this lane's 8 batches (conflict-free LDS broadcast) ----
        v4f xlo = *(const v4f*)&s_x[mo];
        v4f xhi = *(const v4f*)&s_x[mo + 4];
        float xv[8];
        #pragma unroll
        for (int i = 0; i < 4; ++i) { xv[i] = xlo[i]; xv[4 + i] = xhi[i]; }

        // ---- fused GRU cell, fully register-resident ----
        #pragma unroll
        for (int i = 0; i < 8; ++i) {
            const float r  = sigmoidf_(xv[i] * wr + br + acc[0][i] + bhr);
            const float z  = sigmoidf_(xv[i] * wz + bz + acc[1][i] + bhz);
            const float nn = tanhf_   (xv[i] * wn + bn + r * (acc[2][i] + bhn));
            hreg[i] = (1.0f - z) * nn + z * hreg[i];
        }

        // ---- publish updated h as f16 for next step's A operand ----
        #pragma unroll
        for (int i = 0; i < 8; ++i)
            s_h16[(mo + i) * HP + jl] = (_Float16)hreg[i];

        __syncthreads();
    }

    // ---- spill final h to LDS once, then FC: out[b,o] = h[b,:] . fc_w[o,:] + fc_b[o] ----
    #pragma unroll
    for (int i = 0; i < 8; ++i)
        s_h[(mo + i) * H_ + jl] = hreg[i];
    __syncthreads();

    if (tid < BT * O_) {
        const int b = tid >> 2;
        const int o = tid & 3;
        float a = fc_b[o];
        #pragma unroll 4
        for (int j = 0; j < H_; ++j)
            a += s_h[b * H_ + j] * fc_w[o * H_ + j];
        out[(b0 + b) * O_ + o] = a;
    }
}

extern "C" void kernel_launch(void* const* d_in, const int* in_sizes, int n_in,
                              void* d_out, int out_size, void* d_ws, size_t ws_size,
                              hipStream_t stream) {
    const float* x   = (const float*)d_in[0];
    const float* wih = (const float*)d_in[1];
    const float* whh = (const float*)d_in[2];
    const float* bih = (const float*)d_in[3];
    const float* bhh = (const float*)d_in[4];
    const float* fcw = (const float*)d_in[5];
    const float* fcb = (const float*)d_in[6];
    float* out = (float*)d_out;

    dim3 grid(B_ / BT);   // 32 workgroups
    dim3 block(256);      // 8 waves (wave32)
    hipLaunchKernelGGL(gru_persistent_wmma, grid, block, 0, stream,
                       x, wih, whh, bih, bhh, fcw, fcb, out);
}
